// TransformerBlock_82643760710108
// MI455X (gfx1250) — compile-verified, matrix operand order checked by reading, not run on hardware
//
#include <hip/hip_runtime.h>
#include <hip/hip_bf16.h>
#include <math.h>

typedef _Float16 half_t;
typedef __attribute__((ext_vector_type(16))) _Float16 v16h;
typedef __attribute__((ext_vector_type(8)))  _Float16 v8h;
typedef __attribute__((ext_vector_type(4)))  _Float16 v4h;
typedef __attribute__((ext_vector_type(8)))  float    v8f;

constexpr int kB  = 2;
constexpr int kS  = 1024;
constexpr int kH  = 1024;
constexpr int kNH = 16;
constexpr int kHD = 64;
constexpr int kFF = 4096;
constexpr int kE  = 8;
constexpr int kT  = kB * kS;          // 2048 tokens
constexpr float kEPS = 1e-6f;

enum { GEMM_STORE = 0, GEMM_RES_ADD = 1, GEMM_SCALE_SCATTER = 2 };

// Low 32 bits of a generic pointer into LDS = wave-relative LDS byte offset
// (flat-address LDS aperture keeps the offset in ADDR[31:0], ISA 10.2).
__device__ __forceinline__ unsigned lds_offset(const void* p)
{
    return (unsigned)(size_t)p;
}

// ---------------------------------------------------------------------------
// RMSNorm + cast to f16.  One block per token, 256 threads, H = 1024.
// ---------------------------------------------------------------------------
__global__ __launch_bounds__(256)
void rmsnorm_cast_kernel(const float* __restrict__ x, const float* __restrict__ w,
                         half_t* __restrict__ out)
{
    __shared__ float red[256];
    const int t   = blockIdx.x;
    const int tid = threadIdx.x;
    const float* xr = x + (size_t)t * kH;

    float ss = 0.f;
#pragma unroll
    for (int i = 0; i < 4; ++i) {
        float v = xr[tid + i * 256];
        ss += v * v;
    }
    red[tid] = ss;
    __syncthreads();
    for (int s = 128; s > 0; s >>= 1) {
        if (tid < s) red[tid] += red[tid + s];
        __syncthreads();
    }
    const float rinv = rsqrtf(red[0] / (float)kH + kEPS);
#pragma unroll
    for (int i = 0; i < 4; ++i) {
        int j = tid + i * 256;
        out[(size_t)t * kH + j] = (half_t)(xr[j] * rinv * w[j]);
    }
}

// ---------------------------------------------------------------------------
// WMMA GEMM:  C[M,N] (f32) = A[M,K] (f16) * W[N,K]^T (f32->f16 on the fly).
// Block tile 64x128, BK=64; 8 waves as 2(M) x 4(N); wave tile 32x32 =
// 2x2 register-blocked 16x16 WMMA subtiles -> 8 v_wmma per staged tile,
// 2 barriers per 64-wide K stage.
//
// Data movement per stage:
//  - A tile (already f16): GLOBAL_LOAD_ASYNC_TO_LDS_B128 straight into LDS,
//    tracked by ASYNCcnt -> no VGPR round trip; issued first so the W work
//    overlaps it; s_wait_asynccnt 0 before the barrier.
//  - W tile (f32, the BW-critical stream): global_load_b128 clause ->
//    v_cvt_pk_f16_f32 -> ds_store; next K-tile prefetched
//    (global_prefetch_b8).
//
// MoE support (top-2 sparse dispatch, numerically == reference's w_e*y):
//   gatherA  : A row index indirection (token list) for gate/up projections
//   scatterC : C row -> token scatter for the down projection
//   rowscale : per-row gate weight; rows with weight 0 (list padding) are
//              skipped entirely -> no read-modify-write races
//   cntptr   : device count; blocks beyond the count exit before any barrier
// M%64==0 (launch grid), N%128==0, K%64==0 at every call site.
// ---------------------------------------------------------------------------
__global__ __launch_bounds__(256)
void gemm_f16w32_kernel(const half_t* __restrict__ A,
                        const float*  __restrict__ W,
                        float*        __restrict__ C,
                        const float*  __restrict__ res,
                        const float*  __restrict__ rowscale,
                        const int*    __restrict__ gatherA,
                        const int*    __restrict__ scatterC,
                        const int*    __restrict__ cntptr,
                        int M, int N, int Kd, int mode)
{
    __shared__ half_t sA[64 * 64];    // [m][k]
    __shared__ half_t sB[128 * 64];   // [n][k]

    const int tid  = threadIdx.x;
    const int lane = tid & 31;
    const int wave = tid >> 5;
    const int wm   = wave >> 2;              // 0..1
    const int wn   = wave & 3;               // 0..3
    const int nbase = blockIdx.x * 128;
    const int mbase = blockIdx.y * 64;

    const int Mlim = cntptr ? cntptr[0] : M;
    if (mbase >= Mlim) return;

    const v8f vzero = {};
    v8f acc[2][2] = {{vzero, vzero}, {vzero, vzero}};

    // staging indices
    const int ar = tid >> 2;            // 0..63
    const int ac = (tid & 3) * 16;      // 0,16,32,48
    int arow = mbase + ar;
    if (gatherA) arow = gatherA[arow];
    const half_t* aptr = A + (size_t)arow * Kd + ac;
    const unsigned aldst = lds_offset(&sA[ar * 64 + ac]);

    const int bn = tid >> 1;            // 0..127
    const int bc = (tid & 1) * 32;      // 0,32
    const float* wptr = W + (size_t)(nbase + bn) * Kd + bc;

    for (int k0 = 0; k0 < Kd; k0 += 64) {
        // A tile: 64x64 halves, 32 bytes per thread via two async LDS loads
        {
            const half_t* ap = aptr + k0;
            asm volatile("global_load_async_to_lds_b128 %0, %1, off"
                         :: "v"(aldst), "v"(ap) : "memory");
            asm volatile("global_load_async_to_lds_b128 %0, %1, off"
                         :: "v"(aldst + 16u), "v"(ap + 8) : "memory");
        }
        // W tile: 128x64 f32 -> f16, 32 contiguous values per thread
        {
            const float* wp = wptr + k0;
            alignas(16) half_t tmp[32];
#pragma unroll
            for (int i = 0; i < 32; ++i) tmp[i] = (half_t)wp[i];
#pragma unroll
            for (int i = 0; i < 4; ++i)
                *reinterpret_cast<v8h*>(&sB[bn * 64 + bc + i * 8]) =
                    *reinterpret_cast<v8h*>(tmp + i * 8);
            if (k0 + 64 < Kd) __builtin_prefetch(wp + 64, 0, 3);
        }
        asm volatile("s_wait_asynccnt 0" ::: "memory");
        __syncthreads();

        // Two 32-wide WMMA K-steps per staged tile.
        // 16-bit 16x32 A-fragment layout: lane<16: M=lane, halves 0..7 =
        // K 0..7, 8..15 = K 16..23; lane>=16: M=lane-16, K+8 / K+24.
#pragma unroll
        for (int ks = 0; ks < 2; ++ks) {
            const int kk = ks * 32 + (lane >> 4) * 8;
            v16h afr[2], bfr[2];
#pragma unroll
            for (int ms = 0; ms < 2; ++ms) {
                int mr = wm * 32 + ms * 16 + (lane & 15);
                v8h lo = *reinterpret_cast<const v8h*>(&sA[mr * 64 + kk]);
                v8h hi = *reinterpret_cast<const v8h*>(&sA[mr * 64 + kk + 16]);
#pragma unroll
                for (int i = 0; i < 8; ++i) { afr[ms][i] = lo[i]; afr[ms][8 + i] = hi[i]; }
            }
#pragma unroll
            for (int ns = 0; ns < 2; ++ns) {
                int nr = wn * 32 + ns * 16 + (lane & 15);
                v8h lo = *reinterpret_cast<const v8h*>(&sB[nr * 64 + kk]);
                v8h hi = *reinterpret_cast<const v8h*>(&sB[nr * 64 + kk + 16]);
#pragma unroll
                for (int i = 0; i < 8; ++i) { bfr[ns][i] = lo[i]; bfr[ns][8 + i] = hi[i]; }
            }
#pragma unroll
            for (int ms = 0; ms < 2; ++ms)
#pragma unroll
                for (int ns = 0; ns < 2; ++ns)
                    acc[ms][ns] = __builtin_amdgcn_wmma_f32_16x16x32_f16(
                        false, afr[ms], false, bfr[ns], (short)0, acc[ms][ns],
                        false, false);
        }
        __syncthreads();
    }

    // Epilogue. C 16x16 layout: VGPR i -> M = i + 8*(lane>=16), N = lane&15.
#pragma unroll
    for (int ms = 0; ms < 2; ++ms) {
        const int rb = mbase + wm * 32 + ms * 16 + (lane >> 4) * 8;
#pragma unroll
        for (int ns = 0; ns < 2; ++ns) {
            const int col = nbase + wn * 32 + ns * 16 + (lane & 15);
#pragma unroll
            for (int i = 0; i < 8; ++i) {
                int row = rb + i;
                float v = acc[ms][ns][i];
                if (mode == GEMM_SCALE_SCATTER) {
                    float wgt = rowscale[row];
                    if (wgt != 0.f) {
                        int tok = scatterC[row];
                        C[(size_t)tok * N + col] += wgt * v;
                    }
                } else {
                    size_t idx = (size_t)row * N + col;
                    if (mode == GEMM_RES_ADD) v += res[idx];
                    C[idx] = v;
                }
            }
        }
    }
}

// ---------------------------------------------------------------------------
// RoPE (q,k) + 1/sqrt(HD) scale folded into q, cast q/k/v to f16.
// ---------------------------------------------------------------------------
__global__ __launch_bounds__(256)
void rope_cast_kernel(const float* __restrict__ q, const float* __restrict__ k,
                      const float* __restrict__ v,
                      half_t* __restrict__ qh, half_t* __restrict__ kh,
                      half_t* __restrict__ vh)
{
    const int t   = blockIdx.x;
    const int tid = threadIdx.x;
    const int s   = t % kS;
    const float scale = 0.125f;             // 64^-0.5
    const size_t base = (size_t)t * kH;
#pragma unroll
    for (int i = 0; i < 4; ++i) {
        int j  = tid + i * 256;
        int d  = j & (kHD - 1);
        int fi = d & 31;
        float inv_freq = __powf(10000.0f, -(float)(2 * fi) / 64.0f);
        float theta = (float)s * inv_freq;
        float c = __cosf(theta), sn = __sinf(theta);
        int   partner = (d < 32) ? (j + 32) : (j - 32);
        float sgn = (d < 32) ? -1.f : 1.f;
        float qo = q[base + j] * c + sgn * q[base + partner] * sn;
        float ko = k[base + j] * c + sgn * k[base + partner] * sn;
        qh[base + j] = (half_t)(qo * scale);
        kh[base + j] = (half_t)ko;
        vh[base + j] = (half_t)v[base + j];
    }
}

// ---------------------------------------------------------------------------
// Flash attention. Block = 128 threads = 4 waves; wave owns 16 q-rows,
// block covers 64 q-rows of one (batch, head). K/V^T tiles (32 keys) shared
// in LDS. Online softmax through a per-wave LDS score tile; PV accumulated
// in f32 WMMA accumulators with per-row rescale. Output f16 [t][h].
// ---------------------------------------------------------------------------
__global__ __launch_bounds__(128)
void attn_kernel(const half_t* __restrict__ qh, const half_t* __restrict__ kh,
                 const half_t* __restrict__ vh, const float* __restrict__ mask,
                 half_t* __restrict__ out)
{
    __shared__ half_t sK[32 * 64];          // [k_row][d]
    __shared__ half_t sVT[64 * 32];         // [d][k_row]
    __shared__ float  sS[4][16 * 32];
    __shared__ half_t sP[4][16 * 32];
    __shared__ float  sM[4][16], sL[4][16], sC[4][16];

    const int tid  = threadIdx.x;
    const int lane = tid & 31;
    const int wave = tid >> 5;
    const int b    = blockIdx.z;
    const int head = blockIdx.y;
    const int q0   = blockIdx.x * 64 + wave * 16;

    const half_t* qb  = qh + ((size_t)b * kS) * kH + (size_t)head * kHD;
    const half_t* kb_ = kh + ((size_t)b * kS) * kH + (size_t)head * kHD;
    const half_t* vb  = vh + ((size_t)b * kS) * kH + (size_t)head * kHD;

    v16h qa[2];
    {
        int m  = lane & 15;
        int kk = (lane >> 4) * 8;
        const half_t* qr = qb + (size_t)(q0 + m) * kH;
#pragma unroll
        for (int f = 0; f < 2; ++f) {
            v8h lo = *reinterpret_cast<const v8h*>(qr + f * 32 + kk);
            v8h hi = *reinterpret_cast<const v8h*>(qr + f * 32 + kk + 16);
#pragma unroll
            for (int i = 0; i < 8; ++i) { qa[f][i] = lo[i]; qa[f][8 + i] = hi[i]; }
        }
    }

    const v8f vzero = {};
    v8f o[4] = {vzero, vzero, vzero, vzero};
    if (lane < 16) { sM[wave][lane] = -1e30f; sL[wave][lane] = 0.f; }

    for (int kb0 = 0; kb0 < kS; kb0 += 32) {
        {
            int r = tid >> 2;
            int c = (tid & 3) * 16;
            const half_t* krow = kb_ + (size_t)(kb0 + r) * kH + c;
            *reinterpret_cast<v8h*>(&sK[r * 64 + c])     = *reinterpret_cast<const v8h*>(krow);
            *reinterpret_cast<v8h*>(&sK[r * 64 + c + 8]) = *reinterpret_cast<const v8h*>(krow + 8);
            const half_t* vrow = vb + (size_t)(kb0 + r) * kH + c;
            v8h v0 = *reinterpret_cast<const v8h*>(vrow);
            v8h v1 = *reinterpret_cast<const v8h*>(vrow + 8);
#pragma unroll
            for (int i = 0; i < 8; ++i) {
                sVT[(c + i) * 32 + r]     = v0[i];
                sVT[(c + 8 + i) * 32 + r] = v1[i];
            }
        }
        __syncthreads();

        v8f sacc[2] = {vzero, vzero};
#pragma unroll
        for (int n = 0; n < 2; ++n) {
#pragma unroll
            for (int ks = 0; ks < 2; ++ks) {
                int nr = n * 16 + (lane & 15);
                int kk = ks * 32 + (lane >> 4) * 8;
                v16h bf;
                v8h lo = *reinterpret_cast<const v8h*>(&sK[nr * 64 + kk]);
                v8h hi = *reinterpret_cast<const v8h*>(&sK[nr * 64 + kk + 16]);
#pragma unroll
                for (int i = 0; i < 8; ++i) { bf[i] = lo[i]; bf[8 + i] = hi[i]; }
                sacc[n] = __builtin_amdgcn_wmma_f32_16x16x32_f16(
                    false, qa[ks], false, bf, (short)0, sacc[n], false, false);
            }
        }
        {
            int c = lane & 15, rb = (lane >> 4) * 8;
#pragma unroll
            for (int n = 0; n < 2; ++n)
#pragma unroll
                for (int i = 0; i < 8; ++i)
                    sS[wave][(rb + i) * 32 + n * 16 + c] = sacc[n][i];
        }
        __syncthreads();

        if (lane < 16) {
            int r = lane;
            float* srow = &sS[wave][r * 32];
            float mold = sM[wave][r];
            float mx = mold;
#pragma unroll
            for (int c2 = 0; c2 < 32; ++c2) {
                float vv = srow[c2] + mask[(size_t)(q0 + r) * kS + kb0 + c2];
                srow[c2] = vv;
                mx = fmaxf(mx, vv);
            }
            float corr = __expf(mold - mx);
            float l = sL[wave][r] * corr;
#pragma unroll
            for (int c2 = 0; c2 < 32; ++c2) {
                float p = __expf(srow[c2] - mx);
                l += p;
                sP[wave][r * 32 + c2] = (half_t)p;
            }
            sM[wave][r] = mx; sL[wave][r] = l; sC[wave][r] = corr;
        }
        __syncthreads();

        {
            int rb = (lane >> 4) * 8;
            float cr[8];
#pragma unroll
            for (int i = 0; i < 8; ++i) cr[i] = sC[wave][rb + i];
#pragma unroll
            for (int f = 0; f < 4; ++f)
#pragma unroll
                for (int i = 0; i < 8; ++i) o[f][i] *= cr[i];

            v16h pa;
            {
                int m = lane & 15, kk = (lane >> 4) * 8;
                v8h lo = *reinterpret_cast<const v8h*>(&sP[wave][m * 32 + kk]);
                v8h hi = *reinterpret_cast<const v8h*>(&sP[wave][m * 32 + kk + 16]);
#pragma unroll
                for (int i = 0; i < 8; ++i) { pa[i] = lo[i]; pa[8 + i] = hi[i]; }
            }
#pragma unroll
            for (int f = 0; f < 4; ++f) {
                int d  = f * 16 + (lane & 15);
                int kk = (lane >> 4) * 8;
                v16h bf;
                v8h lo = *reinterpret_cast<const v8h*>(&sVT[d * 32 + kk]);
                v8h hi = *reinterpret_cast<const v8h*>(&sVT[d * 32 + kk + 16]);
#pragma unroll
                for (int i = 0; i < 8; ++i) { bf[i] = lo[i]; bf[8 + i] = hi[i]; }
                o[f] = __builtin_amdgcn_wmma_f32_16x16x32_f16(
                    false, pa, false, bf, (short)0, o[f], false, false);
            }
        }
        __syncthreads();
    }

    {
        int c = lane & 15, rb = (lane >> 4) * 8;
#pragma unroll
        for (int f = 0; f < 4; ++f)
#pragma unroll
            for (int i = 0; i < 8; ++i) {
                int r = rb + i;
                float val = o[f][i] / sL[wave][r];
                out[((size_t)b * kS + q0 + r) * kH + head * kHD + f * 16 + c] =
                    (half_t)val;
            }
    }
}

// ---------------------------------------------------------------------------
// MoE bookkeeping reset: cnt[e]=0, token lists / weights zero-filled so
// padded tile rows gather token 0 with weight 0 (skipped at scatter).
// ---------------------------------------------------------------------------
__global__ __launch_bounds__(256)
void moe_fill_kernel(int* __restrict__ cnt, int* __restrict__ list,
                     float* __restrict__ wl)
{
    int i = blockIdx.x * 256 + threadIdx.x;     // covers kE * kT
    if (i < kE) cnt[i] = 0;
    list[i] = 0;
    wl[i] = 0.f;
}

// ---------------------------------------------------------------------------
// Router: logits = x2 @ Wr^T (E=8), wave-per-expert shuffle reduction.
// Thread 0: softmax, first-occurrence top-2 (matches lax.top_k), then
// appends the token to both experts' dispatch lists (int atomics only; the
// final per-token output is independent of list order -> deterministic).
// ---------------------------------------------------------------------------
__global__ __launch_bounds__(256)
void router_kernel(const half_t* __restrict__ x2h, const float* __restrict__ Wr,
                   float* __restrict__ probs, int* __restrict__ cnt,
                   int* __restrict__ list, float* __restrict__ wl)
{
    __shared__ float logits[kE];
    const int t = blockIdx.x, tid = threadIdx.x;
    const int lane = tid & 31, wave = tid >> 5;   // wave == expert
    const half_t* xr = x2h + (size_t)t * kH;
    const float*  wr = Wr + (size_t)wave * kH;
    float acc = 0.f;
    for (int i = lane; i < kH; i += 32) acc += (float)xr[i] * wr[i];
#pragma unroll
    for (int off = 16; off > 0; off >>= 1) acc += __shfl_down(acc, off, 32);
    if (lane == 0) logits[wave] = acc;
    __syncthreads();
    if (tid == 0) {
        float mx = logits[0];
        for (int e = 1; e < kE; ++e) mx = fmaxf(mx, logits[e]);
        float p[kE], sum = 0.f;
        for (int e = 0; e < kE; ++e) { p[e] = __expf(logits[e] - mx); sum += p[e]; }
        for (int e = 0; e < kE; ++e) { p[e] /= sum; probs[(size_t)t * kE + e] = p[e]; }
        int i0 = 0;
        for (int e = 1; e < kE; ++e) if (p[e] > p[i0]) i0 = e;
        int i1 = (i0 == 0) ? 1 : 0;
        for (int e = 0; e < kE; ++e) if (e != i0 && p[e] > p[i1]) i1 = e;
        float wsum = p[i0] + p[i1];
        int p0 = atomicAdd(&cnt[i0], 1);
        list[(size_t)i0 * kT + p0] = t;
        wl[(size_t)i0 * kT + p0] = p[i0] / wsum;
        int p1 = atomicAdd(&cnt[i1], 1);
        list[(size_t)i1 * kT + p1] = t;
        wl[(size_t)i1 * kT + p1] = p[i1] / wsum;
    }
}

// ---------------------------------------------------------------------------
// Load-balance loss: single block, fixed-order tree reduction for the prob
// means; expert counts come from the integer dispatch counters.
// ---------------------------------------------------------------------------
__global__ __launch_bounds__(256)
void lb_loss_kernel(const float* __restrict__ probs, const int* __restrict__ cnt,
                    float* __restrict__ out_lb)
{
    __shared__ float rp[kE][256];
    float ps[kE] = {};
    for (int t = threadIdx.x; t < kT; t += 256)
#pragma unroll
        for (int e = 0; e < kE; ++e) ps[e] += probs[(size_t)t * kE + e];
#pragma unroll
    for (int e = 0; e < kE; ++e) rp[e][threadIdx.x] = ps[e];
    __syncthreads();
    for (int s = 128; s > 0; s >>= 1) {
        if (threadIdx.x < s)
#pragma unroll
            for (int e = 0; e < kE; ++e)
                rp[e][threadIdx.x] += rp[e][threadIdx.x + s];
        __syncthreads();
    }
    if (threadIdx.x == 0) {
        float lb = 0.f;
        for (int e = 0; e < kE; ++e)
            lb += ((float)cnt[e] / (float)kT) * (rp[e][0] / (float)kT);
        out_lb[0] = lb * (float)kE;
    }
}

// ---------------------------------------------------------------------------
// h = silu(gate) * up, cast f16. Bounded to ceil64(count) rows of this
// expert's compact buffer (the rows the down-proj GEMM will actually read).
// ---------------------------------------------------------------------------
__global__ __launch_bounds__(256)
void silu_mul_kernel(const float* __restrict__ g, const float* __restrict__ u,
                     half_t* __restrict__ h, const int* __restrict__ cntptr)
{
    size_t i = (size_t)blockIdx.x * 256 + threadIdx.x;
    int climit = (cntptr[0] + 63) & ~63;
    if (i >= (size_t)climit * kFF) return;
    float gv = g[i];
    float s = gv / (1.f + __expf(-gv));
    h[i] = (half_t)(s * u[i]);
}

// ---------------------------------------------------------------------------
extern "C" void kernel_launch(void* const* d_in, const int* in_sizes, int n_in,
                              void* d_out, int out_size, void* d_ws, size_t ws_size,
                              hipStream_t stream)
{
    (void)in_sizes; (void)n_in; (void)out_size; (void)ws_size;

    const float* hidden = (const float*)d_in[0];
    const float* mask   = (const float*)d_in[1];
    const float* ln1    = (const float*)d_in[2];
    const float* ln2    = (const float*)d_in[3];
    const float* Wq     = (const float*)d_in[4];
    const float* Wk     = (const float*)d_in[5];
    const float* Wv     = (const float*)d_in[6];
    const float* Wo     = (const float*)d_in[7];
    const float* Wr     = (const float*)d_in[8];
    const float* Wg     = (const float*)d_in[9];
    const float* Wu     = (const float*)d_in[10];
    const float* Wd     = (const float*)d_in[11];
    float* out = (float*)d_out;                   // [T*H] hidden + [1] lb_loss

    char* ws = (char*)d_ws;
    auto alloc = [&](size_t bytes) -> char* {
        char* p = ws;
        ws += (bytes + 255) & ~(size_t)255;
        return p;
    };
    half_t* xh    = (half_t*)alloc((size_t)kT * kH * sizeof(half_t));
    float*  qf    = (float*) alloc((size_t)kT * kH * sizeof(float));
    float*  kf    = (float*) alloc((size_t)kT * kH * sizeof(float));
    float*  vf    = (float*) alloc((size_t)kT * kH * sizeof(float));
    half_t* qh    = (half_t*)alloc((size_t)kT * kH * sizeof(half_t));
    half_t* khp   = (half_t*)alloc((size_t)kT * kH * sizeof(half_t));
    half_t* vhp   = (half_t*)alloc((size_t)kT * kH * sizeof(half_t));
    half_t* aoh   = (half_t*)alloc((size_t)kT * kH * sizeof(half_t));
    half_t* x2h   = (half_t*)alloc((size_t)kT * kH * sizeof(half_t));
    float*  probs = (float*) alloc((size_t)kT * kE * sizeof(float));
    int*    cnt   = (int*)   alloc((size_t)kE * sizeof(int));
    int*    list  = (int*)   alloc((size_t)kE * kT * sizeof(int));
    float*  wl    = (float*) alloc((size_t)kE * kT * sizeof(float));
    float*  gate  = (float*) alloc((size_t)kT * kFF * sizeof(float));
    float*  up    = (float*) alloc((size_t)kT * kFF * sizeof(float));
    half_t* hsel  = (half_t*)alloc((size_t)kT * kFF * sizeof(half_t));

    const dim3 blk256(256), blk128(128);
    const dim3 gH(kH / 128, kT / 64);      // GEMM grids (N/128, M/64)
    const dim3 gFF(kFF / 128, kT / 64);

    // 1. RMSNorm + cast
    rmsnorm_cast_kernel<<<kT, blk256, 0, stream>>>(hidden, ln1, xh);

    // 2. Q/K/V projections
    gemm_f16w32_kernel<<<gH, blk256, 0, stream>>>(xh, Wq, qf, nullptr, nullptr,
        nullptr, nullptr, nullptr, kT, kH, kH, GEMM_STORE);
    gemm_f16w32_kernel<<<gH, blk256, 0, stream>>>(xh, Wk, kf, nullptr, nullptr,
        nullptr, nullptr, nullptr, kT, kH, kH, GEMM_STORE);
    gemm_f16w32_kernel<<<gH, blk256, 0, stream>>>(xh, Wv, vf, nullptr, nullptr,
        nullptr, nullptr, nullptr, kT, kH, kH, GEMM_STORE);

    // 3. RoPE + scale + cast
    rope_cast_kernel<<<kT, blk256, 0, stream>>>(qf, kf, vf, qh, khp, vhp);

    // 4. Attention
    attn_kernel<<<dim3(kS / 64, kNH, kB), blk128, 0, stream>>>(qh, khp, vhp, mask, aoh);

    // 5. Output projection + residual -> hidden (d_out)
    gemm_f16w32_kernel<<<gH, blk256, 0, stream>>>(aoh, Wo, out, hidden, nullptr,
        nullptr, nullptr, nullptr, kT, kH, kH, GEMM_RES_ADD);

    // 6. Second RMSNorm
    rmsnorm_cast_kernel<<<kT, blk256, 0, stream>>>(out, ln2, x2h);

    // 7. Router: reset dispatch state, route, LB loss
    moe_fill_kernel<<<(kE * kT) / 256, blk256, 0, stream>>>(cnt, list, wl);
    router_kernel<<<kT, blk256, 0, stream>>>(x2h, Wr, probs, cnt, list, wl);
    lb_loss_kernel<<<1, blk256, 0, stream>>>(probs, cnt, out + (size_t)kT * kH);

    // 8. MoE experts, top-2 sparse dispatch:
    //    gate/up gather tokens from the expert list; down-proj scatters
    //    w_e * y back to d_out (weight-0 padding rows skipped).
    for (int e = 0; e < kE; ++e) {
        const float* Wge = Wg + (size_t)e * kFF * kH;
        const float* Wue = Wu + (size_t)e * kFF * kH;
        const float* Wde = Wd + (size_t)e * kH * kFF;
        const int*   le  = list + (size_t)e * kT;
        const float* we  = wl + (size_t)e * kT;
        gemm_f16w32_kernel<<<gFF, blk256, 0, stream>>>(x2h, Wge, gate, nullptr,
            nullptr, le, nullptr, cnt + e, kT, kFF, kH, GEMM_STORE);
        gemm_f16w32_kernel<<<gFF, blk256, 0, stream>>>(x2h, Wue, up, nullptr,
            nullptr, le, nullptr, cnt + e, kT, kFF, kH, GEMM_STORE);
        silu_mul_kernel<<<(kT * kFF) / 256, blk256, 0, stream>>>(gate, up, hsel, cnt + e);
        gemm_f16w32_kernel<<<gH, blk256, 0, stream>>>(hsel, Wde, out, nullptr,
            we, nullptr, le, cnt + e, kT, kH, kFF, GEMM_SCALE_SCATTER);
    }
}
